// GAT_62483184222291
// MI455X (gfx1250) — matrix operand order repaired and checked by reading, not compiled, not run on hardware
//
#include <hip/hip_runtime.h>
#include <hip/hip_bf16.h>

// ---------------- problem constants (match reference) ----------------
#define NN    100000          // nodes
#define EE    1600000         // directed edges (before self loops)
#define ETOT  (EE + NN)       // edges incl self loops
#define DIN   128
#define HID   32
#define HEADS 4
#define D1    128             // HID*HEADS
#define NGR   64
#define NEG_SLOPE 0.2f

typedef __attribute__((ext_vector_type(16))) _Float16 v16h;
typedef __attribute__((ext_vector_type(8)))  float    v8f;

// ---------------- ordered-float <-> uint monotone map (for atomicMax) ----
__device__ __forceinline__ unsigned f2ord(float f) {
  unsigned u = __float_as_uint(f);
  return (u & 0x80000000u) ? ~u : (u | 0x80000000u);
}
__device__ __forceinline__ float ord2f(unsigned u) {
  unsigned v = (u & 0x80000000u) ? (u & 0x7fffffffu) : ~u;
  return __uint_as_float(v);
}

__device__ __forceinline__ float elu1(float v) {
  return v > 0.f ? v : (__expf(v) - 1.f);
}
__device__ __forceinline__ float lrelu(float v) {
  return v > 0.f ? v : NEG_SLOPE * v;
}

// ---------------- generic zero fill (u32 words) ----------------
__global__ void k_zero(unsigned* __restrict__ p, int nwords) {
  int t = blockIdx.x * blockDim.x + threadIdx.x;
  int stride = gridDim.x * blockDim.x;
  for (int i = t; i < nwords; i += stride) p[i] = 0u;
}

// ---------------- pre-swizzle W (f32 [128][128] row-major, W[o][k]) into
// per-lane v16h B-fragments for v_wmma_f32_16x16x32_f16.
// B is 32x16 (KxN) with Bt[k][n] = W[n_col][k]; lane layout assumed:
//   n = lane&15 (column), hi = lane>>4, half j<8 -> K = kbase + 8*hi + j, j>=8 -> K = kbase + 8*hi + 16 + (j-8)
// frag index = (tileN*4 + kstep)*32 + lane ----------------
__global__ void k_preswizzle(const float* __restrict__ W, v16h* __restrict__ out) {
  int t = blockIdx.x * blockDim.x + threadIdx.x;   // 0..1023
  if (t >= 8 * 4 * 32) return;
  int lane = t & 31;
  int s    = (t >> 5) & 3;   // kstep
  int tile = t >> 7;         // tileN
  int n  = lane & 15;
  int hi = lane >> 4;
  int ocol  = tile * 16 + n;
  int kbase = s * 32 + hi * 8;
  v16h v;
#pragma unroll
  for (int j = 0; j < 16; ++j) v[j] = (_Float16)W[ocol * 128 + kbase + (j < 8 ? j : j + 8)];
  out[(tile * 4 + s) * 32 + lane] = v;
}

// ---------------- WMMA GEMM: Hout[N][128] = Hin[N][128] @ W^T ----------------
// one wave per 16x16 output tile, K loop of 4x wmma_f32_16x16x32_f16.
// A (16x32 f16) lane layout (ISA 7.12.2): m = lane&15, hi = lane>>4,
//   halves 0..7  -> K = kbase + 8*hi + j
//   halves 8..15 -> K = kbase + 8*hi + 16 + (j-8)
__global__ __launch_bounds__(32) void k_gemm(const float* __restrict__ Hin,
                                             const v16h* __restrict__ wfrag,
                                             float* __restrict__ Hout) {
  int tileM = blockIdx.x;            // 0..6249
  int tileN = blockIdx.y;            // 0..7
  int lane  = threadIdx.x;           // 0..31
  int m  = lane & 15;
  int hi = lane >> 4;
  int row = tileM * 16 + m;
  const float* arow = Hin + (long)row * 128;

  v8f acc = {};
#pragma unroll
  for (int s = 0; s < 4; ++s) {
    int k0 = s * 32 + hi * 8;
    const float4* p0 = (const float4*)(arow + k0);
    const float4* p1 = (const float4*)(arow + k0 + 16);
    float4 x0 = p0[0], x1 = p0[1];
    float4 y0 = p1[0], y1 = p1[1];
    v16h a;
    a[0]  = (_Float16)x0.x; a[1]  = (_Float16)x0.y;
    a[2]  = (_Float16)x0.z; a[3]  = (_Float16)x0.w;
    a[4]  = (_Float16)x1.x; a[5]  = (_Float16)x1.y;
    a[6]  = (_Float16)x1.z; a[7]  = (_Float16)x1.w;
    a[8]  = (_Float16)y0.x; a[9]  = (_Float16)y0.y;
    a[10] = (_Float16)y0.z; a[11] = (_Float16)y0.w;
    a[12] = (_Float16)y1.x; a[13] = (_Float16)y1.y;
    a[14] = (_Float16)y1.z; a[15] = (_Float16)y1.w;
    v16h b = wfrag[(tileN * 4 + s) * 32 + lane];
    acc = __builtin_amdgcn_wmma_f32_16x16x32_f16(false, a, false, b,
                                                 (short)0, acc, false, false);
  }
  // D layout: col = tileN*16 + (lane&15); row = tileM*16 + 8*hi + r
  int col   = tileN * 16 + m;
  int rbase = tileM * 16 + hi * 8;
#pragma unroll
  for (int r = 0; r < 8; ++r)
    Hout[(long)(rbase + r) * 128 + col] = acc[r];
}

// ---------------- per-node attention head scores ----------------
__global__ void k_node_alpha(const float* __restrict__ hT,
                             const float* __restrict__ aw_s,
                             const float* __restrict__ aw_d,
                             float* __restrict__ asrc,
                             float* __restrict__ adst) {
  int t = blockIdx.x * blockDim.x + threadIdx.x;
  if (t >= NN * HEADS) return;
  int n = t >> 2, h = t & 3;
  const float* hp = hT + (long)n * 128 + h * 32;
  const float* ws = aw_s + h * 32;
  const float* wd = aw_d + h * 32;
  float s = 0.f, d = 0.f;
#pragma unroll
  for (int c = 0; c < 32; ++c) {
    float v = hp[c];
    s += v * ws[c];
    d += v * wd[c];
  }
  asrc[t] = s;
  adst[t] = d;
}

__device__ __forceinline__ void edge_sd(const int* __restrict__ ei, int t,
                                        int& s, int& d) {
  if (t < EE) { s = ei[t]; d = ei[EE + t]; }
  else        { s = d = t - EE; }
}

// ---------------- edge pass A: scores + segment max ----------------
__global__ void k_edge_a(const int* __restrict__ ei,
                         const float* __restrict__ asrc,
                         const float* __restrict__ adst,
                         float* __restrict__ ebuf,
                         unsigned* __restrict__ emax) {
  int t = blockIdx.x * blockDim.x + threadIdx.x;
  if (t >= ETOT) return;
  int s, d;
  edge_sd(ei, t, s, d);
  float4 av = *(const float4*)(asrc + (long)s * 4);
  float4 dv = *(const float4*)(adst + (long)d * 4);
  float4 e;
  e.x = lrelu(av.x + dv.x);
  e.y = lrelu(av.y + dv.y);
  e.z = lrelu(av.z + dv.z);
  e.w = lrelu(av.w + dv.w);
  *(float4*)(ebuf + (long)t * 4) = e;
  unsigned* mp = emax + (long)d * 4;
  atomicMax(mp + 0, f2ord(e.x));
  atomicMax(mp + 1, f2ord(e.y));
  atomicMax(mp + 2, f2ord(e.z));
  atomicMax(mp + 3, f2ord(e.w));
}

// ---------------- edge pass B: exp(e - max) + segment sum ----------------
__global__ void k_edge_b(const int* __restrict__ ei,
                         float* __restrict__ ebuf,
                         const unsigned* __restrict__ emax,
                         float* __restrict__ denom) {
  int t = blockIdx.x * blockDim.x + threadIdx.x;
  if (t >= ETOT) return;
  int s, d;
  edge_sd(ei, t, s, d);
  float4 e = *(const float4*)(ebuf + (long)t * 4);
  const unsigned* mp = emax + (long)d * 4;
  float4 ex;
  ex.x = __expf(e.x - ord2f(mp[0]));
  ex.y = __expf(e.y - ord2f(mp[1]));
  ex.z = __expf(e.z - ord2f(mp[2]));
  ex.w = __expf(e.w - ord2f(mp[3]));
  *(float4*)(ebuf + (long)t * 4) = ex;
  float* dp = denom + (long)d * 4;
  atomicAdd(dp + 0, ex.x);
  atomicAdd(dp + 1, ex.y);
  atomicAdd(dp + 2, ex.z);
  atomicAdd(dp + 3, ex.w);
}

// ---------------- edge pass C: weighted scatter, 1 wave / edge ----------------
__global__ void k_edge_c(const int* __restrict__ ei,
                         const float* __restrict__ hT,
                         const float* __restrict__ ebuf,
                         const float* __restrict__ denom,
                         float* __restrict__ hAgg) {
  int tid  = blockIdx.x * blockDim.x + threadIdx.x;
  int edge = tid >> 5;
  int lane = tid & 31;
  if (edge >= ETOT) return;
  int s, d;
  edge_sd(ei, edge, s, d);
  int f0   = lane * 4;        // 4 features per lane
  int head = lane >> 3;       // 32 features per head
  float alpha = ebuf[(long)edge * 4 + head] / denom[(long)d * 4 + head];
  float4 hv = *(const float4*)(hT + (long)s * 128 + f0);
  float* op = hAgg + (long)d * 128 + f0;
  atomicAdd(op + 0, hv.x * alpha);
  atomicAdd(op + 1, hv.y * alpha);
  atomicAdd(op + 2, hv.z * alpha);
  atomicAdd(op + 3, hv.w * alpha);
}

// ---------------- finalize layers 1/2: +bias, ELU ----------------
__global__ void k_fin12(const float* __restrict__ hAgg,
                        const float* __restrict__ bias,
                        float* __restrict__ hIn) {
  int t = blockIdx.x * blockDim.x + threadIdx.x;
  if (t >= NN * 128) return;
  int f = t & 127;
  hIn[t] = elu1(hAgg[t] + bias[f]);
}

// ---------------- finalize layer 3: head-mean, +bias, ELU ----------------
__global__ void k_fin3(const float* __restrict__ hAgg,
                       const float* __restrict__ b3,
                       float* __restrict__ h3) {
  int t = blockIdx.x * blockDim.x + threadIdx.x;
  if (t >= NN * 32) return;
  int n = t >> 5, c = t & 31;
  const float* p = hAgg + (long)n * 128;
  float v = 0.25f * (p[c] + p[32 + c] + p[64 + c] + p[96 + c]) + b3[c];
  h3[t] = elu1(v);
}

// ---------------- readout + graph pooling ----------------
__global__ void k_readout(const float* __restrict__ h3,
                          const float* __restrict__ lin_w,
                          const float* __restrict__ lin_b,
                          const int* __restrict__ batch,
                          float* __restrict__ psum,
                          float* __restrict__ pcnt) {
  int t = blockIdx.x * blockDim.x + threadIdx.x;
  if (t >= NN) return;
  const float* hp = h3 + (long)t * 32;
  float y = lin_b[0];
#pragma unroll
  for (int c = 0; c < 32; ++c) y += hp[c] * lin_w[c];
  int g = batch[t];
  atomicAdd(&psum[g], y);
  atomicAdd(&pcnt[g], 1.f);
}

__global__ void k_divide(const float* __restrict__ psum,
                         const float* __restrict__ pcnt,
                         float* __restrict__ out) {
  int t = blockIdx.x * blockDim.x + threadIdx.x;
  if (t >= NGR) return;
  out[t] = psum[t] / pcnt[t];
}

// ---------------- workspace layout (bytes, 256-aligned) ----------------
#define OFF_HIN   0ULL                               // N*128*4 = 51,200,000
#define OFF_HT    (OFF_HIN  + 51200000ULL)
#define OFF_AGG   (OFF_HT   + 51200000ULL)           // zeroed each layer
#define OFF_EMAX  (OFF_AGG  + 51200000ULL)           // N*4*4 (contiguous w/ AGG)
#define OFF_DEN   (OFF_EMAX + 1600000ULL)            // N*4*4 (contiguous)
#define OFF_AS    (OFF_DEN  + 1600000ULL)
#define OFF_AD    (OFF_AS   + 1600000ULL)
#define OFF_EB    (OFF_AD   + 1600000ULL)            // ETOT*4*4 = 27,200,000
#define OFF_H3    (OFF_EB   + 27200000ULL)           // N*32*4 = 12,800,000
#define OFF_POOL  (OFF_H3   + 12800000ULL)           // psum 64 + pcnt 64 (+pad)
#define OFF_WF    (OFF_POOL + 1024ULL)               // 3 * 32768 B f16 fragments

extern "C" void kernel_launch(void* const* d_in, const int* in_sizes, int n_in,
                              void* d_out, int out_size, void* d_ws, size_t ws_size,
                              hipStream_t stream) {
  const float* x     = (const float*)d_in[0];
  const int*   ei    = (const int*)  d_in[1];
  const int*   batch = (const int*)  d_in[2];
  const float* W[3]  = { (const float*)d_in[3], (const float*)d_in[7],  (const float*)d_in[11] };
  const float* As[3] = { (const float*)d_in[4], (const float*)d_in[8],  (const float*)d_in[12] };
  const float* Ad[3] = { (const float*)d_in[5], (const float*)d_in[9],  (const float*)d_in[13] };
  const float* B[3]  = { (const float*)d_in[6], (const float*)d_in[10], (const float*)d_in[14] };
  const float* lin_w = (const float*)d_in[15];
  const float* lin_b = (const float*)d_in[16];

  char* ws = (char*)d_ws;
  float*    hIn   = (float*)   (ws + OFF_HIN);
  float*    hT    = (float*)   (ws + OFF_HT);
  float*    hAgg  = (float*)   (ws + OFF_AGG);
  unsigned* emax  = (unsigned*)(ws + OFF_EMAX);
  float*    denom = (float*)   (ws + OFF_DEN);
  float*    asrc  = (float*)   (ws + OFF_AS);
  float*    adst  = (float*)   (ws + OFF_AD);
  float*    ebuf  = (float*)   (ws + OFF_EB);
  float*    h3    = (float*)   (ws + OFF_H3);
  float*    psum  = (float*)   (ws + OFF_POOL);
  float*    pcnt  = psum + 64;
  v16h*     wf    = (v16h*)    (ws + OFF_WF);   // 1024 frags per layer

  // pre-swizzle all three weight matrices into WMMA B-fragments
  for (int l = 0; l < 3; ++l)
    k_preswizzle<<<4, 256, 0, stream>>>(W[l], wf + (size_t)l * 1024);

  const int zwords = (51200000 + 1600000 + 1600000) / 4;  // hAgg + emax + denom
  dim3 gGemm(NN / 16, 8);

  for (int l = 0; l < 3; ++l) {
    const float* Hin = (l == 0) ? x : hIn;
    k_zero<<<2048, 256, 0, stream>>>((unsigned*)hAgg, zwords);
    k_gemm<<<gGemm, 32, 0, stream>>>(Hin, wf + (size_t)l * 1024, hT);
    k_node_alpha<<<(NN * HEADS + 255) / 256, 256, 0, stream>>>(hT, As[l], Ad[l], asrc, adst);
    k_edge_a<<<(ETOT + 255) / 256, 256, 0, stream>>>(ei, asrc, adst, ebuf, emax);
    k_edge_b<<<(ETOT + 255) / 256, 256, 0, stream>>>(ei, ebuf, emax, denom);
    k_edge_c<<<(ETOT * 32 + 255) / 256, 256, 0, stream>>>(ei, hT, ebuf, denom, hAgg);
    if (l < 2) k_fin12<<<(NN * 128 + 255) / 256, 256, 0, stream>>>(hAgg, B[l], hIn);
    else       k_fin3 <<<(NN * 32  + 255) / 256, 256, 0, stream>>>(hAgg, B[l], h3);
  }

  k_zero<<<1, 256, 0, stream>>>((unsigned*)psum, 256);
  k_readout<<<(NN + 255) / 256, 256, 0, stream>>>(h3, lin_w, lin_b, batch, psum, pcnt);
  k_divide<<<1, 64, 0, stream>>>(psum, pcnt, (float*)d_out);
}
